// VectorQuantize_9706626089877
// MI455X (gfx1250) — matrix operand order repaired and checked by reading, not compiled, not run on hardware
//
#include <hip/hip_runtime.h>

#define D       512
#define NROWS   16384
#define NC      8192
#define QELEMS  (NROWS * D)     // 8388608
#define SA      520             // padded LDS row stride in bf16 units (1040B -> bank-conflict-free b128)

typedef __bf16        v16bf __attribute__((ext_vector_type(16)));
typedef float         v8f   __attribute__((ext_vector_type(8)));
typedef unsigned int  u32x4 __attribute__((ext_vector_type(4)));

union FragU { u32x4 u[2]; v16bf v; };   // trivially-constructible (ext vectors only)

__device__ __forceinline__ unsigned short f2bf(float f) {
  unsigned int u = __float_as_uint(f);
  unsigned int r = u + 0x7FFFu + ((u >> 16) & 1u);   // round-to-nearest-even
  return (unsigned short)(r >> 16);
}
__device__ __forceinline__ float bf2f(unsigned short h) {
  return __uint_as_float(((unsigned int)h) << 16);
}

__device__ __forceinline__ v8f wmma_bf16(v16bf a, v16bf b, v8f c) {
  // 8 args: (neg_a, A, neg_b, B, c_mod, C, reuse_a, reuse_b)
  return __builtin_amdgcn_wmma_f32_16x16x32_bf16(false, a, false, b, (short)0, c, false, false);
}

// Async global -> LDS copy, 16B per lane, tracked with ASYNCcnt.
__device__ __forceinline__ void async_copy_b128(unsigned ldsOff, const void* gaddr) {
  asm volatile("global_load_async_to_lds_b128 %0, %1, off"
               :: "v"(ldsOff), "v"((unsigned long long)(uintptr_t)gaddr)
               : "memory");
}
__device__ __forceinline__ void wait_asynccnt0() {
  asm volatile("s_wait_asynccnt 0" ::: "memory");
}

// ---------------------------------------------------------------------------
// Kernel 1: e2[c] = ||embed_c||^2 ; split embed into bf16 hi/lo arrays.
// grid = NC/8 blocks x 256 threads (one wave32 per code row)
// ---------------------------------------------------------------------------
__global__ __launch_bounds__(256) void vq_prep_kernel(
    const float* __restrict__ embed, float* __restrict__ e2,
    unsigned short* __restrict__ ehi, unsigned short* __restrict__ elo) {
  const int row  = blockIdx.x * 8 + (threadIdx.x >> 5);
  const int lane = threadIdx.x & 31;
  const float4* src = reinterpret_cast<const float4*>(embed + (size_t)row * D);
  float ss = 0.f;
#pragma unroll
  for (int j = 0; j < 4; ++j) {
    const int c4 = lane + j * 32;           // 128 float4 per row, coalesced
    float4 v = src[c4];
    float vv[4] = {v.x, v.y, v.z, v.w};
    unsigned short hs[4], ls[4];
#pragma unroll
    for (int t = 0; t < 4; ++t) {
      ss += vv[t] * vv[t];
      unsigned short h = f2bf(vv[t]);
      hs[t] = h;
      ls[t] = f2bf(vv[t] - bf2f(h));
    }
    const size_t o = (size_t)row * D + (size_t)c4 * 4;
    *reinterpret_cast<ushort4*>(ehi + o) = make_ushort4(hs[0], hs[1], hs[2], hs[3]);
    *reinterpret_cast<ushort4*>(elo + o) = make_ushort4(ls[0], ls[1], ls[2], ls[3]);
  }
#pragma unroll
  for (int m = 16; m >= 1; m >>= 1) ss += __shfl_xor(ss, m, 32);
  if (lane == 0) e2[row] = ss;
}

// ---------------------------------------------------------------------------
// Kernel 2: fused bf16x3 WMMA GEMM (x . e^T) + running argmin of e2 - 2*dot.
// grid = NROWS/64 blocks x 256 threads (8 waves = 4 row-groups x 2 col-groups)
// B chunks double-buffered in LDS via GLOBAL_LOAD_ASYNC_TO_LDS_B128.
// ---------------------------------------------------------------------------
__global__ __launch_bounds__(256, 1) void vq_argmin_kernel(
    const float* __restrict__ x, const unsigned short* __restrict__ ehi,
    const unsigned short* __restrict__ elo, const float* __restrict__ e2,
    int* __restrict__ ind) {
  __shared__ unsigned short Ah[64 * SA];
  __shared__ unsigned short Al[64 * SA];
  __shared__ unsigned short Bh[2][32 * SA];
  __shared__ unsigned short Bl[2][32 * SA];
  __shared__ float redV[128];
  __shared__ int   redI[128];

  const int tid     = threadIdx.x;
  const int lane    = tid & 31;
  const int w       = tid >> 5;
  const int rg      = w >> 1;           // 0..3 : 16-row group
  const int cg      = w & 1;            // 0..1 : 16-col group within chunk
  const int rowBase = blockIdx.x * 64;

  // Issue async fill of B buffer p with embed chunk `chunk` (hi+lo).
  auto stageAsync = [&](int chunk, int p) {
#pragma unroll
    for (int j = 0; j < 8; ++j) {
      const int i  = tid + j * 256;          // 2048 16B-chunks per array
      const int r  = i >> 6;
      const int c8 = i & 63;
      const size_t g = (size_t)(chunk * 32 + r) * D + (size_t)c8 * 8;
      const int    o = r * SA + c8 * 8;
      async_copy_b128((unsigned)(uintptr_t)(&Bh[p][o]), ehi + g);
      async_copy_b128((unsigned)(uintptr_t)(&Bl[p][o]), elo + g);
    }
  };

  // Stage x tile (64 x 512 f32) -> bf16 hi/lo in LDS (done once, reused 256x).
  for (int i = tid; i < 64 * 128; i += 256) {
    const int r  = i >> 7;
    const int c4 = i & 127;
    float4 v = reinterpret_cast<const float4*>(x + (size_t)(rowBase + r) * D)[c4];
    float vv[4] = {v.x, v.y, v.z, v.w};
    unsigned short hs[4], ls[4];
#pragma unroll
    for (int t = 0; t < 4; ++t) {
      unsigned short h = f2bf(vv[t]);
      hs[t] = h;
      ls[t] = f2bf(vv[t] - bf2f(h));
    }
    const int o = r * SA + c4 * 4;
    *reinterpret_cast<ushort4*>(Ah + o) = make_ushort4(hs[0], hs[1], hs[2], hs[3]);
    *reinterpret_cast<ushort4*>(Al + o) = make_ushort4(ls[0], ls[1], ls[2], ls[3]);
  }

  stageAsync(0, 0);   // prologue fill of buffer 0

  float bestV[8];
  int   bestI[8];
#pragma unroll
  for (int v = 0; v < 8; ++v) { bestV[v] = 3.4e38f; bestI[v] = 0; }

  // Wave32 fragment addressing per ISA 16-bit A(16x32)/B(32x16) layouts.
  const int aRow = rg * 16 + (lane & 15);
  const int bRow = cg * 16 + (lane & 15);
  const int aSel = (lane < 16) ? 0 : 8;    // A: K 0..7 / 8..15, second chunk +16
  const int bSel = (lane < 16) ? 0 : 8;    // B: K 0..7 / 8..15, second chunk +16

  for (int chunk = 0; chunk < NC / 32; ++chunk) {
    const int p     = chunk & 1;
    const int cbase = chunk * 32;

    wait_asynccnt0();   // this wave's fill of buffer p complete
    __syncthreads();    // all waves' fills visible; prev compute reads of p^1 done

    if (chunk + 1 < NC / 32) stageAsync(chunk + 1, p ^ 1);

    // Hoist e2 load so its latency hides under the WMMA K-loop.
    const int   col = cbase + cg * 16 + (lane & 15);
    const float e2v = e2[col];

    const unsigned short* BhP = Bh[p];
    const unsigned short* BlP = Bl[p];

    v8f acc = {0.f, 0.f, 0.f, 0.f, 0.f, 0.f, 0.f, 0.f};
#pragma unroll 4
    for (int kb = 0; kb < D; kb += 32) {
      FragU ah, al, bh, bl;
      const unsigned short* pa = Ah + aRow * SA + kb + aSel;
      ah.u[0] = *reinterpret_cast<const u32x4*>(pa);
      ah.u[1] = *reinterpret_cast<const u32x4*>(pa + 16);
      const unsigned short* qa = Al + aRow * SA + kb + aSel;
      al.u[0] = *reinterpret_cast<const u32x4*>(qa);
      al.u[1] = *reinterpret_cast<const u32x4*>(qa + 16);
      const unsigned short* pb = BhP + bRow * SA + kb + bSel;
      bh.u[0] = *reinterpret_cast<const u32x4*>(pb);
      bh.u[1] = *reinterpret_cast<const u32x4*>(pb + 16);
      const unsigned short* qb = BlP + bRow * SA + kb + bSel;
      bl.u[0] = *reinterpret_cast<const u32x4*>(qb);
      bl.u[1] = *reinterpret_cast<const u32x4*>(qb + 16);
      acc = wmma_bf16(ah.v, bh.v, acc);   // hi*hi
      acc = wmma_bf16(ah.v, bl.v, acc);   // hi*lo
      acc = wmma_bf16(al.v, bh.v, acc);   // lo*hi
    }

#pragma unroll
    for (int v = 0; v < 8; ++v) {
      const float s = e2v - 2.0f * acc[v];          // ||e||^2 - 2 x.e
      if (s < bestV[v]) { bestV[v] = s; bestI[v] = col; }  // strict <: first-occurrence ties
    }
  }

  // Min across the 16 lanes of each half-wave (rows v / v+8), index tie-break.
#pragma unroll
  for (int m = 8; m >= 1; m >>= 1) {
#pragma unroll
    for (int v = 0; v < 8; ++v) {
      const float ov = __shfl_xor(bestV[v], m, 32);
      const int   oi = __shfl_xor(bestI[v], m, 32);
      if (ov < bestV[v] || (ov == bestV[v] && oi < bestI[v])) { bestV[v] = ov; bestI[v] = oi; }
    }
  }
  __syncthreads();
  if ((lane & 15) == 0) {
    const int half = lane >> 4;
#pragma unroll
    for (int v = 0; v < 8; ++v) {
      const int r = rg * 16 + half * 8 + v;
      redV[r * 2 + cg] = bestV[v];
      redI[r * 2 + cg] = bestI[v];
    }
  }
  __syncthreads();
  if (tid < 64) {
    const float v0 = redV[tid * 2 + 0], v1 = redV[tid * 2 + 1];
    const int   i0 = redI[tid * 2 + 0], i1 = redI[tid * 2 + 1];
    ind[rowBase + tid] = (v1 < v0 || (v1 == v0 && i1 < i0)) ? i1 : i0;
  }
}

// ---------------------------------------------------------------------------
// Kernel 3: gather quantize, write index-as-float, per-row partial MSE.
// grid = NROWS blocks x 256 threads (one row per block).
// ---------------------------------------------------------------------------
__global__ __launch_bounds__(256) void vq_gather_kernel(
    const float* __restrict__ x, const float* __restrict__ embed,
    const int* __restrict__ ind, float* __restrict__ qout,
    float* __restrict__ indOut, float* __restrict__ partial) {
  __shared__ float sred[8];
  const int row  = blockIdx.x;
  const int tid  = threadIdx.x;
  const int lane = tid & 31, w = tid >> 5;
  const int idx  = ind[row];
  float2 q  = reinterpret_cast<const float2*>(embed + (size_t)idx * D)[tid];
  float2 xv = reinterpret_cast<const float2*>(x + (size_t)row * D)[tid];
  reinterpret_cast<float2*>(qout + (size_t)row * D)[tid] = q;   // quantize_st == quantize
  const float d0 = q.x - xv.x, d1 = q.y - xv.y;
  float ss = d0 * d0 + d1 * d1;
#pragma unroll
  for (int m = 16; m >= 1; m >>= 1) ss += __shfl_xor(ss, m, 32);
  if (lane == 0) sred[w] = ss;
  __syncthreads();
  if (tid == 0) {
    float t = 0.f;
#pragma unroll
    for (int i = 0; i < 8; ++i) t += sred[i];
    partial[row]  = t;
    indOut[row]   = (float)idx;
  }
}

// ---------------------------------------------------------------------------
// Kernel 4: deterministic final reduction -> commitment loss scalar.
// ---------------------------------------------------------------------------
__global__ __launch_bounds__(256) void vq_loss_kernel(
    const float* __restrict__ partial, float* __restrict__ lossOut) {
  __shared__ float sred[8];
  const int tid = threadIdx.x, lane = tid & 31, w = tid >> 5;
  float s = 0.f;
  for (int i = tid; i < NROWS; i += 256) s += partial[i];
#pragma unroll
  for (int m = 16; m >= 1; m >>= 1) s += __shfl_xor(s, m, 32);
  if (lane == 0) sred[w] = s;
  __syncthreads();
  if (tid == 0) {
    float t = 0.f;
#pragma unroll
    for (int i = 0; i < 8; ++i) t += sred[i];
    lossOut[0] = t / (float)QELEMS;   // mean over B*N*D, COMMITMENT_WEIGHT = 1.0
  }
}

extern "C" void kernel_launch(void* const* d_in, const int* in_sizes, int n_in,
                              void* d_out, int out_size, void* d_ws, size_t ws_size,
                              hipStream_t stream) {
  const float* x     = (const float*)d_in[0];   // [8,2048,512] f32
  const float* embed = (const float*)d_in[1];   // [1,8192,512] f32
  float* out = (float*)d_out;                   // quantize | ind(as f32) | loss

  char* ws = (char*)d_ws;                       // ~16.2 MB used
  float*          e2      = (float*)(ws);
  unsigned short* ehi     = (unsigned short*)(ws + (32u << 10));
  unsigned short* elo     = (unsigned short*)(ws + (32u << 10) + (size_t)NC * D * 2);
  int*            ind     = (int*)  (ws + (32u << 10) + (size_t)NC * D * 4);
  float*          partial = (float*)(ws + (32u << 10) + (size_t)NC * D * 4 + (size_t)NROWS * 4);

  vq_prep_kernel  <<<NC / 8,     256, 0, stream>>>(embed, e2, ehi, elo);
  vq_argmin_kernel<<<NROWS / 64, 256, 0, stream>>>(x, ehi, elo, e2, ind);
  vq_gather_kernel<<<NROWS,      256, 0, stream>>>(x, embed, ind, out, out + QELEMS, partial);
  vq_loss_kernel  <<<1,          256, 0, stream>>>(partial, out + QELEMS + NROWS);
}
